// VectorizedExpertMLP_10952166604895
// MI455X (gfx1250) — matrix operand order repaired and checked by reading, not compiled, not run on hardware
//
#include <hip/hip_runtime.h>
#include <hip/hip_bf16.h>
#include <stdint.h>

typedef __bf16 bf16;
typedef __attribute__((ext_vector_type(16))) __bf16 v16bf;
typedef __attribute__((ext_vector_type(4)))  __bf16 v4bf;
typedef __attribute__((ext_vector_type(8)))  float  v8f;
typedef __attribute__((ext_vector_type(4)))  float  v4f;

#define T_TOK 4096
#define DDIM  1024
#define IDIM  2048
#define NEXP  8
#define TOPK  2

#define BM   128
#define BN   128
#define BK   32
#define PADK 40   // 40 halves = 80B = 20 dwords row stride: conflict-free (20*m mod 64 distinct), 16B aligned

// wt[e*T + t] = sum_k (selected_experts[t][k] == e) ? routing_weights[t][k] : 0
__global__ __launch_bounds__(256) void route_weights(const int* __restrict__ se,
                                                     const float* __restrict__ rw,
                                                     float* __restrict__ wt) {
  int t = blockIdx.x * 256 + threadIdx.x;
  if (t >= T_TOK) return;
  int   e0 = se[t * TOPK + 0], e1 = se[t * TOPK + 1];
  float r0 = rw[t * TOPK + 0], r1 = rw[t * TOPK + 1];
#pragma unroll
  for (int e = 0; e < NEXP; ++e) {
    wt[e * T_TOK + t] = (e0 == e ? r0 : 0.f) + (e1 == e ? r1 : 0.f);
  }
}

// Tiled bf16 WMMA GEMM, double-buffered LDS.
// A path: f32 -> reg-stage + packed cvt (GEMM1), or bf16 -> GLOBAL_LOAD_ASYNC_TO_LDS_B128 (GEMM2).
// EPI 0: Hout = bf16(silu(acc))   EPI 1: Cout += acc*wt   EPI 2: Cout = acc*wt
template <int EPI, bool ABF16>
__global__ __launch_bounds__(256) void moe_gemm(
    const void*  __restrict__ Aany,
    const float* __restrict__ B,       // f32 row-major [Kdim x ldb]
    bf16*        __restrict__ Hout,
    float*       __restrict__ Cout,
    const float* __restrict__ wt_e,
    int Kdim, int lda, int ldb, int Ncols)
{
  __shared__ bf16 As[2][BM][PADK];
  __shared__ bf16 Bs[2][BN][PADK];   // transposed: Bs[b][n][k]

  const int tid     = threadIdx.x;
  const int rowBase = blockIdx.y * BM;
  const int colBase = blockIdx.x * BN;

  const int wave = tid >> 5;
  const int lane = tid & 31;
  const int lm   = lane & 15;
  const int hi   = lane >> 4;
  const int m0w  = (wave & 3) * 32;   // 4 waves along M
  const int n0w  = (wave >> 2) * 64;  // 2 waves along N

  // A staging coords: f32 path (float4 chunks) and async path (b128 = 8 halves)
  const int arow = tid >> 3;          // f32: 0..31, +i*32
  const int acol = (tid & 7) * 4;
  const int xrow = tid >> 2;          // async: 0..63, +i*64
  const int xcol = (tid & 3) * 8;
  // B: thread owns a 4x4 (k x n) block
  const int bk   = (tid >> 5) * 4;
  const int bn   = (tid & 31) * 4;

  const float* Af  = (const float*)Aany;
  const bf16*  Ab  = (const bf16*)Aany;
  const float* aPf = Af + (size_t)(rowBase + arow) * lda + acol;
  const bf16*  aPh = Ab + (size_t)(rowBase + xrow) * lda + xcol;
  const float* bP  = B  + (size_t)bk * ldb + colBase + bn;

  // low 32 bits of the generic pointer == LDS byte offset (ISA aperture rule)
  const unsigned aLds0 = (unsigned)(uintptr_t)&As[0][xrow][xcol];
  const unsigned aLds1 = (unsigned)(uintptr_t)&As[1][xrow][xcol];

  v8f acc[2][4] = {};
  v4f aV[4]; v4f bV[4];

  auto loadGlobal = [&](int b) {
    if constexpr (ABF16) {
      // async DMA straight into the idle LDS buffer; tracked by ASYNCcnt
      unsigned base = b ? aLds1 : aLds0;
#pragma unroll
      for (int i = 0; i < 2; ++i) {
        unsigned lds = base + (unsigned)(i * 64 * PADK * 2);
        unsigned long long g =
            (unsigned long long)(uintptr_t)(aPh + (size_t)(i * 64) * lda);
        asm volatile("global_load_async_to_lds_b128 %0, %1, off"
                     :: "v"(lds), "v"(g) : "memory");
      }
      aPh += BK;
    } else {
#pragma unroll
      for (int i = 0; i < 4; ++i)
        aV[i] = *(const v4f*)(aPf + (size_t)(i * 32) * lda);
      aPf += BK;
    }
#pragma unroll
    for (int i = 0; i < 4; ++i) bV[i] = *(const v4f*)(bP + (size_t)i * ldb);
    bP += (size_t)BK * ldb;
  };

  auto storeLDS = [&](int b) {
    if constexpr (!ABF16) {
#pragma unroll
      for (int i = 0; i < 4; ++i) {   // A: packed 8B stores
        v4bf h;
        h[0] = (bf16)aV[i][0]; h[1] = (bf16)aV[i][1];
        h[2] = (bf16)aV[i][2]; h[3] = (bf16)aV[i][3];
        *(v4bf*)&As[b][arow + i * 32][acol] = h;
      }
    }
#pragma unroll
    for (int j = 0; j < 4; ++j) {     // B: transpose 4x4 block in regs, packed stores
      v4bf h;
#pragma unroll
      for (int i = 0; i < 4; ++i) h[i] = (bf16)bV[i][j];
      *(v4bf*)&Bs[b][bn + j][bk] = h;
    }
    if constexpr (ABF16)              // A tile landing via async copies
      asm volatile("s_wait_asynccnt 0x0" ::: "memory");
  };

  auto compute = [&](int b) {
    v16bf afr[2], bfr[4];
#pragma unroll
    for (int tm = 0; tm < 2; ++tm) {
      int row = m0w + tm * 16 + lm;
#pragma unroll
      for (int q = 0; q < 16; ++q) afr[tm][q] = As[b][row][hi * 16 + q];
    }
#pragma unroll
    for (int tn = 0; tn < 4; ++tn) {
      int col = n0w + tn * 16 + lm;
#pragma unroll
      for (int q = 0; q < 16; ++q) bfr[tn][q] = Bs[b][col][hi * 16 + q];
    }
#pragma unroll
    for (int tm = 0; tm < 2; ++tm)
#pragma unroll
      for (int tn = 0; tn < 4; ++tn)
        acc[tm][tn] = __builtin_amdgcn_wmma_f32_16x16x32_bf16(
            false, afr[tm], false, bfr[tn], (short)0, acc[tm][tn], false, false);
  };

  // ---- pipeline: 1 barrier/stage; global/async transfers overlap the WMMA stream
  loadGlobal(0);
  storeLDS(0);
  __syncthreads();
  const int nK = Kdim / BK;
  for (int kt = 0; kt < nK; ++kt) {
    bool more = (kt + 1 < nK);
    if (more) loadGlobal((kt + 1) & 1);  // buffer was released by previous barrier
    compute(kt & 1);
    if (more) {
      storeLDS((kt + 1) & 1);
      __syncthreads();
    }
  }

  // ---- epilogue: C/D layout -> row = r + 8*hi, col = lm per 16x16 tile
#pragma unroll
  for (int tm = 0; tm < 2; ++tm) {
#pragma unroll
    for (int tn = 0; tn < 4; ++tn) {
#pragma unroll
      for (int r = 0; r < 8; ++r) {
        int grow = rowBase + m0w + tm * 16 + hi * 8 + r;
        int gcol = colBase + n0w + tn * 16 + lm;
        float v = acc[tm][tn][r];
        if constexpr (EPI == 0) {
          float s = __fdividef(v, 1.f + __expf(-v));   // fast silu (rcp path)
          Hout[(size_t)grow * Ncols + gcol] = (bf16)s;
        } else if constexpr (EPI == 1) {
          size_t o = (size_t)grow * Ncols + gcol;
          Cout[o] += v * wt_e[grow];
        } else {
          size_t o = (size_t)grow * Ncols + gcol;
          Cout[o] = v * wt_e[grow];
        }
      }
    }
  }
}

extern "C" void kernel_launch(void* const* d_in, const int* in_sizes, int n_in,
                              void* d_out, int out_size, void* d_ws, size_t ws_size,
                              hipStream_t stream) {
  const float* x  = (const float*)d_in[0];   // [4096, 1024]
  const float* rw = (const float*)d_in[1];   // [4096, 2]
  const int*   se = (const int*)d_in[2];     // [4096, 2] (int32: JAX x64 disabled)
  const float* w1 = (const float*)d_in[3];   // [8, 1024, 2048]
  const float* w2 = (const float*)d_in[4];   // [8, 2048, 1024]
  float* out = (float*)d_out;                // [4096, 1024]

  char*  ws = (char*)d_ws;
  float* wt = (float*)ws;                                          // 8*4096 f32 = 128 KB
  bf16*  hb = (bf16*)(ws + (size_t)NEXP * T_TOK * sizeof(float));  // 4096*2048 bf16 = 16 MB

  route_weights<<<(T_TOK + 255) / 256, 256, 0, stream>>>(se, rw, wt);

  dim3 blk(256);
  dim3 g1(IDIM / BN, T_TOK / BM);   // 16 x 32 workgroups
  dim3 g2(DDIM / BN, T_TOK / BM);   //  8 x 32 workgroups
  for (int e = 0; e < NEXP; ++e) {
    const float* w1e = w1 + (size_t)e * DDIM * IDIM;
    const float* w2e = w2 + (size_t)e * IDIM * DDIM;
    const float* wte = wt + (size_t)e * T_TOK;
    // h = silu(x @ w1e) in bf16
    moe_gemm<0, false><<<g1, blk, 0, stream>>>(x,  w1e, hb, nullptr, nullptr,
                                               DDIM, DDIM, IDIM, IDIM);
    // out (=|+=) (h @ w2e) * wt[e][t]  — A operand streams via async-to-LDS
    if (e == 0)
      moe_gemm<2, true><<<g2, blk, 0, stream>>>(hb, w2e, nullptr, out, wte,
                                                IDIM, IDIM, DDIM, DDIM);
    else
      moe_gemm<1, true><<<g2, blk, 0, stream>>>(hb, w2e, nullptr, out, wte,
                                                IDIM, IDIM, DDIM, DDIM);
  }
}
